// ProtoPNetConv_15092515078250
// MI455X (gfx1250) — compile-verified, matrix operand order checked by reading, not run on hardware
//
#include <hip/hip_runtime.h>
#include <hip/hip_bf16.h>
#include <math.h>

typedef __attribute__((ext_vector_type(16))) _Float16 v16h;
typedef __attribute__((ext_vector_type(8)))  float    v8f;

#define BATCH 64
#define SEQ   512
#define ENC   1024
#define NPROTO 100
#define PSIZE 4
#define NCOL  400          // NPROTO * PSIZE
#define NTILES 25          // NCOL / 16
#define KSTEPS 32          // ENC / 32
#define LDSPITCH 36        // floats per staged row (32 + 4 pad, 16B-aligned)

// ---------------------------------------------------------------------------
// Kernel 1: pack protolayer [P,E,K] f32 -> WMMA-B-fragment-ordered f16 buffer
// layout: Wb[((t*KSTEPS + ks)*32 + lane)*16 + i]
//   n = t*16 + (lane&15), kb = (lane&16)?8:0, e = ks*32 + kb + (i&7) + ((i>>3)<<4)
// ---------------------------------------------------------------------------
__launch_bounds__(256)
__global__ void ppn_pack_b(const float* __restrict__ proto, _Float16* __restrict__ Wb) {
  int idx = blockIdx.x * 256 + threadIdx.x;           // 0 .. 409599
  int i    = idx & 15;
  int lane = (idx >> 4) & 31;
  int ks   = (idx >> 9) & 31;
  int t    = idx >> 14;                               // 0 .. 24
  int n  = t * 16 + (lane & 15);
  int kb = (lane & 16) ? 8 : 0;
  int e  = ks * 32 + kb + (i & 7) + ((i >> 3) << 4);
  int p  = n >> 2;
  int kk = n & 3;
  Wb[idx] = (_Float16)proto[(size_t)p * (ENC * PSIZE) + (size_t)e * PSIZE + kk];
}

// ---------------------------------------------------------------------------
// Kernel 2: p2[p] = sum over (E,K) of proto^2
// ---------------------------------------------------------------------------
__launch_bounds__(256)
__global__ void ppn_p2(const float* __restrict__ proto, float* __restrict__ p2) {
  __shared__ float red[256];
  int p = blockIdx.x;
  const float* row = proto + (size_t)p * (ENC * PSIZE);
  float s = 0.f;
  for (int i = threadIdx.x; i < ENC * PSIZE; i += 256) { float v = row[i]; s += v * v; }
  red[threadIdx.x] = s; __syncthreads();
  for (int w = 128; w > 0; w >>= 1) {
    if (threadIdx.x < w) red[threadIdx.x] += red[threadIdx.x + w];
    __syncthreads();
  }
  if (threadIdx.x == 0) p2[p] = red[0];
}

// ---------------------------------------------------------------------------
// Kernel 3: s2[b,s] = sum_e embedding[b,s,e]^2   (contiguous coalesced row)
// ---------------------------------------------------------------------------
__launch_bounds__(256)
__global__ void ppn_s2(const float* __restrict__ emb, float* __restrict__ s2) {
  __shared__ float red[256];
  int s = blockIdx.x, b = blockIdx.y;
  const float* row = emb + ((size_t)b * SEQ + s) * ENC;
  float acc = 0.f;
  for (int i = threadIdx.x; i < ENC; i += 256) { float v = row[i]; acc += v * v; }
  red[threadIdx.x] = acc; __syncthreads();
  for (int w = 128; w > 0; w >>= 1) {
    if (threadIdx.x < w) red[threadIdx.x] += red[threadIdx.x + w];
    __syncthreads();
  }
  if (threadIdx.x == 0) s2[(size_t)b * SEQ + s] = red[0];
}

// ---------------------------------------------------------------------------
// Kernel 4: batched GEMM  G[b, s, n] = sum_e emb[b,s,e] * Wpack[e,n]
// grid = (NTILES, BATCH); block = 256 (8 waves). Block tile M=512 (whole S),
// N=16. Each wave owns 4 m-tiles accumulated against one shared B fragment.
// Double-buffered LDS: while computing on buffer k, the 16 async
// GLOBAL_LOAD_ASYNC_TO_LDS_B128 per thread for K-step k+1 are in flight;
// s_wait_asynccnt 0x10 requires only the previous batch to have landed
// (async loads complete in order), overlapping DMA with WMMA.
// ---------------------------------------------------------------------------
__launch_bounds__(256)
__global__ void ppn_gemm(const float* __restrict__ X, const _Float16* __restrict__ Wb,
                         float* __restrict__ G) {
  __shared__ float As[2][SEQ * LDSPITCH];           // 2 x 73,728 bytes
  const int nt = blockIdx.x;           // n-tile 0..24
  const int b  = blockIdx.y;           // batch
  const int tid  = threadIdx.x;
  const int wave = tid >> 5;
  const int lane = tid & 31;
  const float* xbase = X + (size_t)b * SEQ * ENC;

  const int lrow = wave * 64 + (lane & 15);       // first A row of this wave's tiles
  const int kb   = (lane & 16) ? 8 : 0;           // A/B fragment K base
  const _Float16* bptr = Wb + (((size_t)nt * KSTEPS * 32) + lane) * 16;

  // per-thread staging coordinates: 512 rows x 8 float4 = 4096 float4, 16/thread
  auto stage = [&](int ksrc, int buf) {
#pragma unroll
    for (int j = 0; j < 16; ++j) {
      int f = tid + 256 * j;
      int r = f >> 3, c4 = f & 7;
      const float* src = xbase + (size_t)r * ENC + ksrc * 32 + c4 * 4;
      unsigned lds = (unsigned)(uintptr_t)(&As[buf][r * LDSPITCH + c4 * 4]);
      asm volatile("global_load_async_to_lds_b128 %0, %1, off"
                   :: "v"(lds), "v"(src) : "memory");
    }
  };

  v8f acc[4] = {{}, {}, {}, {}};

  stage(0, 0);                         // prologue fill of buffer 0

  for (int ks = 0; ks < KSTEPS; ++ks) {
    const int cur = ks & 1;
    __syncthreads();                   // all waves done reading buf[cur^1]
    if (ks + 1 < KSTEPS) {
      stage(ks + 1, cur ^ 1);          // overlap next fill with this compute
      asm volatile("s_wait_asynccnt 0x10" ::: "memory");  // prev batch landed
    } else {
      asm volatile("s_wait_asynccnt 0x0" ::: "memory");   // drain
    }
    __syncthreads();                   // publish buf[cur] to all waves

    // B fragment: pre-packed, one contiguous 32B load per lane; prefetch next
    v16h bf = *(const v16h*)(bptr + (size_t)ks * (32 * 16));
    if (ks + 1 < KSTEPS)
      __builtin_prefetch(bptr + (size_t)(ks + 1) * (32 * 16), 0, 3);

    // 4 m-tiles against the shared B fragment
#pragma unroll
    for (int t = 0; t < 4; ++t) {
      const float* arow = &As[cur][(lrow + t * 16) * LDSPITCH + kb];
      float4 f0 = *(const float4*)(arow + 0);
      float4 f1 = *(const float4*)(arow + 4);
      float4 f2 = *(const float4*)(arow + 16);
      float4 f3 = *(const float4*)(arow + 20);
      v16h a;
      a[0]  = (_Float16)f0.x; a[1]  = (_Float16)f0.y; a[2]  = (_Float16)f0.z; a[3]  = (_Float16)f0.w;
      a[4]  = (_Float16)f1.x; a[5]  = (_Float16)f1.y; a[6]  = (_Float16)f1.z; a[7]  = (_Float16)f1.w;
      a[8]  = (_Float16)f2.x; a[9]  = (_Float16)f2.y; a[10] = (_Float16)f2.z; a[11] = (_Float16)f2.w;
      a[12] = (_Float16)f3.x; a[13] = (_Float16)f3.y; a[14] = (_Float16)f3.z; a[15] = (_Float16)f3.w;
      acc[t] = __builtin_amdgcn_wmma_f32_16x16x32_f16(
          false, a, false, bf, (short)0, acc[t], false, false);
    }
  }

  // C/D layout: VGPR r, lanes 0-15 -> M=r, lanes 16-31 -> M=r+8; N = lane&15
  const int msub = wave * 64 + ((lane & 16) ? 8 : 0);
  const int ncol = nt * 16 + (lane & 15);
#pragma unroll
  for (int t = 0; t < 4; ++t) {
    float* gout = G + ((size_t)b * SEQ + msub + t * 16) * NCOL + ncol;
#pragma unroll
    for (int r = 0; r < 8; ++r) gout[(size_t)r * NCOL] = acc[t][r];
  }
}

// ---------------------------------------------------------------------------
// Kernel 5: distances + per-(b,p) min.
// grid = (NPROTO, BATCH); block 256.
//   d = p/25 + 1, L = 512 - 3d
//   dist[b,pp,l] = sqrt(relu( sum_k s2[b,l+kd] - 2*sum_k G[b,l+kd,p*4+k] + p2[p] ))
// ---------------------------------------------------------------------------
__launch_bounds__(256)
__global__ void ppn_dist(const float* __restrict__ G, const float* __restrict__ s2,
                         const float* __restrict__ p2, float* __restrict__ out) {
  __shared__ float red[256];
  const int p = blockIdx.x;
  const int b = blockIdx.y;
  const int dg = p / 25;               // dilation group 0..3
  const int d  = dg + 1;
  const int L  = SEQ - 3 * d;
  size_t base = 6400;
  for (int g = 0; g < dg; ++g) base += (size_t)BATCH * 25 * (SEQ - 3 * (g + 1));
  const int pp = p - dg * 25;
  float* distout = out + base + ((size_t)b * 25 + pp) * L;

  const float* grow  = G  + (size_t)b * SEQ * NCOL + (size_t)p * 4;
  const float* s2row = s2 + (size_t)b * SEQ;
  const float p2v = p2[p];

  float lmin = 3.402823466e38f;
  for (int l = threadIdx.x; l < L; l += 256) {
    float xp = 0.f, x2s = 0.f;
#pragma unroll
    for (int k = 0; k < 4; ++k) {
      int r = l + k * d;
      xp  += grow[(size_t)r * NCOL + k];
      x2s += s2row[r];
    }
    float v = x2s - 2.f * xp + p2v;
    v = v > 0.f ? v : 0.f;
    float dist = sqrtf(v);
    distout[l] = dist;
    lmin = fminf(lmin, dist);
  }
  red[threadIdx.x] = lmin; __syncthreads();
  for (int w = 128; w > 0; w >>= 1) {
    if (threadIdx.x < w) red[threadIdx.x] = fminf(red[threadIdx.x], red[threadIdx.x + w]);
    __syncthreads();
  }
  if (threadIdx.x == 0) out[(size_t)b * NPROTO + p] = red[0];
}

// ---------------------------------------------------------------------------
// Kernel 6: class_out[b,c] = sum_p pd[b,p] * fc_w[c,p]
// ---------------------------------------------------------------------------
__launch_bounds__(128)
__global__ void ppn_fc(const float* __restrict__ fc_w, float* __restrict__ out,
                       size_t class_base) {
  int t = threadIdx.x;                 // 0..127 -> (b, c)
  int b = t >> 1, c = t & 1;
  const float* pd = out + (size_t)b * NPROTO;
  float s = 0.f;
  for (int p = 0; p < NPROTO; ++p) s += pd[p] * fc_w[c * NPROTO + p];
  out[class_base + (size_t)b * 2 + c] = s;
}

// ---------------------------------------------------------------------------
extern "C" void kernel_launch(void* const* d_in, const int* in_sizes, int n_in,
                              void* d_out, int out_size, void* d_ws, size_t ws_size,
                              hipStream_t stream) {
  const float* emb   = (const float*)d_in[0];   // [64,512,1024]
  const float* proto = (const float*)d_in[1];   // [100,1024,4]
  const float* fc_w  = (const float*)d_in[2];   // [2,100]
  float* out = (float*)d_out;

  // workspace layout
  char* ws = (char*)d_ws;
  _Float16* Wb = (_Float16*)ws;                                   //   819,200 B
  float* G  = (float*)(ws + 819200);                              // 52,428,800 B
  float* s2 = (float*)(ws + 819200 + 52428800);                   //    131,072 B
  float* p2 = s2 + (size_t)BATCH * SEQ;                           //        400 B

  ppn_pack_b<<<1600, 256, 0, stream>>>(proto, Wb);
  ppn_p2<<<NPROTO, 256, 0, stream>>>(proto, p2);
  ppn_s2<<<dim3(SEQ, BATCH), 256, 0, stream>>>(emb, s2);
  ppn_gemm<<<dim3(NTILES, BATCH), 256, 0, stream>>>(emb, Wb, G);
  ppn_dist<<<dim3(NPROTO, BATCH), 256, 0, stream>>>(G, s2, p2, out);

  size_t class_base = 6400;
  for (int g = 0; g < 4; ++g) class_base += (size_t)BATCH * 25 * (SEQ - 3 * (g + 1));
  ppn_fc<<<1, 128, 0, stream>>>(fc_w, out, class_base);
}
